// RNNModel_22007412425012
// MI455X (gfx1250) — matrix operand order repaired and checked by reading, not compiled, not run on hardware
//
#include <hip/hip_runtime.h>
#include <cmath>

#define T_SEQ  512
#define BATCH  256
#define IN_DIM 300
#define HID    256

// LDS row strides padded so stride_dwords % 8 == 4: the 16 laneLo lanes of a
// b128 LDS access start at distinct multiples of 4 banks -> conflict-free.
#define KPAD    328   // xproj W_ih rows (>= 320), 164 dw = 4*41
#define WSTRIDE 264   // scan W_hh / h-buffer rows (>= 256), 132 dw = 4*33

typedef __attribute__((ext_vector_type(16))) _Float16 v16h;
typedef __attribute__((ext_vector_type(8)))  _Float16 v8h;
typedef __attribute__((ext_vector_type(8)))  float    v8f;

union Frag16 {              // 16 f16 = two 16B halves (b128 LDS ops)
    v16h v;
    v8h  h[2];
    _Float16 e[16];
};

// CDNA5 hardware tanh (32-bit transcendental VALU op)
__device__ __forceinline__ float tanh_hw(float x) {
    float r;
    asm("v_tanh_f32 %0, %1" : "=v"(r) : "v"(x));
    return r;
}

// ---------------------------------------------------------------------------
// Phase 1: xp[t*B+b, h] = sum_i x[t,b,i] * W_ih[h,i] + b_ih[h] + b_hh[h]
// M = T*B = 131072, N = 256, K = 300 (padded to 320). Output stored f16.
// ---------------------------------------------------------------------------
__global__ __launch_bounds__(256) void k_xproj(
    const float* __restrict__ x, const float* __restrict__ W_ih,
    const float* __restrict__ b_ih, const float* __restrict__ b_hh,
    _Float16* __restrict__ xp)
{
    extern __shared__ __attribute__((aligned(32))) char smem_raw[];
    _Float16* Wl = (_Float16*)smem_raw;              // [256][KPAD] f16

    const int tid    = threadIdx.x;
    const int lane   = tid & 31;
    const int wave   = tid >> 5;
    const int laneLo = lane & 15;
    const int laneHi = lane >> 4;

    // Stage W_ih -> LDS f16 (row n contiguous in k == B-frag layout),
    // zero-pad K 300..KPAD. float4 global loads (1200B rows are 16B aligned).
    {
        const float4* src = (const float4*)(W_ih + (size_t)tid * IN_DIM);
        _Float16*     dst = Wl + (size_t)tid * KPAD;
        for (int k4 = 0; k4 < IN_DIM / 4; ++k4) {
            const float4 v = src[k4];
            dst[k4 * 4 + 0] = (_Float16)v.x;
            dst[k4 * 4 + 1] = (_Float16)v.y;
            dst[k4 * 4 + 2] = (_Float16)v.z;
            dst[k4 * 4 + 3] = (_Float16)v.w;
        }
        for (int k = IN_DIM; k < KPAD; ++k) dst[k] = (_Float16)0.0f;
    }
    __syncthreads();

    const int nBase = (wave & 3) * 64;
    float bias[4];
#pragma unroll
    for (int nt = 0; nt < 4; ++nt)
        bias[nt] = b_ih[nBase + nt * 16 + laneLo] + b_hh[nBase + nt * 16 + laneLo];

    // B fragments: one base, offsets nt*16*KPAD + kt*32 (immediates)
    const _Float16* wbase = Wl + (size_t)(nBase + laneLo) * KPAD + laneHi * 8;

    const int MCHUNKS = (T_SEQ * BATCH) / 32;
    for (int mc = blockIdx.x; mc < MCHUNKS; mc += gridDim.x) {
        const int mBase = mc * 32 + (wave >> 2) * 16;
        // A stream base: row (mBase+laneLo), k offset laneHi*8
        const float* xbase = x + (size_t)(mBase + laneLo) * IN_DIM + laneHi * 8;
        // D stream base: row (mBase+laneHi*8), col (nBase+laneLo)
        _Float16* obase = xp + (size_t)(mBase + laneHi * 8) * HID + nBase + laneLo;

        // Bias folded into C-init (one accumulation, added exactly once)
        v8f acc[4];
#pragma unroll
        for (int nt = 0; nt < 4; ++nt)
#pragma unroll
            for (int r = 0; r < 8; ++r) acc[nt][r] = bias[nt];

        // 9 full K-tiles (K = 0..287), partially unrolled to cap VGPR use
#pragma unroll 3
        for (int kt = 0; kt < 9; ++kt) {
            Frag16 a;
            const float4 p0 = ((const float4*)(xbase + kt * 32))[0];
            const float4 p1 = ((const float4*)(xbase + kt * 32))[1];
            const float4 p2 = ((const float4*)(xbase + kt * 32 + 16))[0];
            const float4 p3 = ((const float4*)(xbase + kt * 32 + 16))[1];
            a.e[0]  = (_Float16)p0.x; a.e[1]  = (_Float16)p0.y;
            a.e[2]  = (_Float16)p0.z; a.e[3]  = (_Float16)p0.w;
            a.e[4]  = (_Float16)p1.x; a.e[5]  = (_Float16)p1.y;
            a.e[6]  = (_Float16)p1.z; a.e[7]  = (_Float16)p1.w;
            a.e[8]  = (_Float16)p2.x; a.e[9]  = (_Float16)p2.y;
            a.e[10] = (_Float16)p2.z; a.e[11] = (_Float16)p2.w;
            a.e[12] = (_Float16)p3.x; a.e[13] = (_Float16)p3.y;
            a.e[14] = (_Float16)p3.z; a.e[15] = (_Float16)p3.w;
#pragma unroll
            for (int nt = 0; nt < 4; ++nt) {
                Frag16 b;
                b.h[0] = *(const v8h*)(wbase + nt * 16 * KPAD + kt * 32);
                b.h[1] = *(const v8h*)(wbase + nt * 16 * KPAD + kt * 32 + 16);
                acc[nt] = __builtin_amdgcn_wmma_f32_16x16x32_f16(
                    false, a.v, false, b.v, (short)0, acc[nt], false, false);
            }
        }
        { // tail tile kt=9: K 288..319, valid only < 300 (scalar, guarded)
            Frag16 a;
#pragma unroll
            for (int i = 0; i < 8; ++i) {
                const int c = 288 + laneHi * 8 + i;
                a.e[i] = (c < IN_DIM) ? (_Float16)xbase[288 + i] : (_Float16)0.0f;
            }
#pragma unroll
            for (int i = 0; i < 8; ++i) a.e[8 + i] = (_Float16)0.0f;  // K>=304
#pragma unroll
            for (int nt = 0; nt < 4; ++nt) {
                Frag16 b;
                b.h[0] = *(const v8h*)(wbase + nt * 16 * KPAD + 9 * 32);
                b.h[1] = *(const v8h*)(wbase + nt * 16 * KPAD + 9 * 32 + 16);
                acc[nt] = __builtin_amdgcn_wmma_f32_16x16x32_f16(
                    false, a.v, false, b.v, (short)0, acc[nt], false, false);
            }
        }

        // D: VGPR r -> row r / r+8, col lane%16; store via base + immediates
#pragma unroll
        for (int nt = 0; nt < 4; ++nt)
#pragma unroll
            for (int r = 0; r < 8; ++r)
                obase[(size_t)r * HID + nt * 16] = (_Float16)acc[nt][r];
    }
}

// ---------------------------------------------------------------------------
// Phase 2: recurrent scan. 16 blocks x 16 batch rows (independent rows ->
// no inter-block sync). W_hh resident in LDS f16; h double-buffered in LDS,
// all LDS rows padded to WSTRIDE for conflict-free b128 access.
// Software-pipelined xp loads (regs); xpv folded into the even-K chain init.
// ---------------------------------------------------------------------------
__global__ __launch_bounds__(256) void k_scan(
    const _Float16* __restrict__ xp, const float* __restrict__ W_hh,
    const float* __restrict__ W_fc, const float* __restrict__ b_fc,
    float* __restrict__ out)
{
    extern __shared__ __attribute__((aligned(32))) char smem_raw2[];
    _Float16* Wl  = (_Float16*)smem_raw2;            // [256][WSTRIDE] f16 W_hh
    _Float16* hbA = Wl  + (size_t)HID * WSTRIDE;     // [16][WSTRIDE] f16
    _Float16* hbB = hbA + (size_t)16 * WSTRIDE;      // [16][WSTRIDE] f16

    const int tid    = threadIdx.x;
    const int lane   = tid & 31;
    const int wave   = tid >> 5;
    const int laneLo = lane & 15;
    const int laneHi = lane >> 4;

    { // stage W_hh -> LDS f16 (1024B global rows, float4 loads)
        const float4* src = (const float4*)(W_hh + (size_t)tid * HID);
        _Float16*     dst = Wl + (size_t)tid * WSTRIDE;
        for (int k4 = 0; k4 < HID / 4; ++k4) {
            const float4 v = src[k4];
            dst[k4 * 4 + 0] = (_Float16)v.x;
            dst[k4 * 4 + 1] = (_Float16)v.y;
            dst[k4 * 4 + 2] = (_Float16)v.z;
            dst[k4 * 4 + 3] = (_Float16)v.w;
        }
    }
    for (int i = tid; i < 16 * WSTRIDE; i += 256) hbA[i] = (_Float16)0.0f; // h0=0
    __syncthreads();

    const int bRow0 = blockIdx.x * 16;   // this block's batch rows
    const int nBase = wave * 32;         // 2 N-tiles per wave
    const int rowD  = laneHi * 8;        // C/D row offset for hi lanes

    // Loop-invariant lane bases (hot addressing = base + immediate)
    const _Float16* wb = Wl + (size_t)(nBase + laneLo) * WSTRIDE + laneHi * 8;
    const _Float16* aA = hbA + (size_t)laneLo * WSTRIDE + laneHi * 8;
    const _Float16* aB = hbB + (size_t)laneLo * WSTRIDE + laneHi * 8;
    _Float16* oA = hbA + (size_t)rowD * WSTRIDE + nBase + laneLo;
    _Float16* oB = hbB + (size_t)rowD * WSTRIDE + nBase + laneLo;

    // xp stream: element (t*256 + bRow0 + rowD + r)*256 + nBase + nt*16 + laneLo
    const _Float16* xprow = xp + ((size_t)bRow0 + rowD) * HID + nBase + laneLo;

    // Prime the xp pipeline for t = 0
    float xpv[2][8];
#pragma unroll
    for (int nt = 0; nt < 2; ++nt)
#pragma unroll
        for (int r = 0; r < 8; ++r)
            xpv[nt][r] = (float)xprow[(size_t)r * HID + nt * 16];

    auto step = [&](const _Float16* aIn, _Float16* hOut, int t) {
        // Issue next step's xp loads first; they retire under the WMMA chain.
        float xn[2][8];
        const _Float16* xnrow = xprow + (size_t)BATCH * HID;
        if (t + 1 < T_SEQ) {
#pragma unroll
            for (int nt = 0; nt < 2; ++nt)
#pragma unroll
                for (int r = 0; r < 8; ++r)
                    xn[nt][r] = (float)xnrow[(size_t)r * HID + nt * 16];
        }

        // 4 independent chains: (nt=0/1) x (even/odd K), depth 4 each.
        // Even chain starts from xp (pipelined regs), odd chain from zero.
        v8f acc0[2], acc1[2];
#pragma unroll
        for (int nt = 0; nt < 2; ++nt)
#pragma unroll
            for (int r = 0; r < 8; ++r) { acc0[nt][r] = xpv[nt][r]; acc1[nt][r] = 0.0f; }

#pragma unroll
        for (int kt = 0; kt < 8; ++kt) {
            Frag16 a;
            a.h[0] = *(const v8h*)(aIn + kt * 32);
            a.h[1] = *(const v8h*)(aIn + kt * 32 + 16);
#pragma unroll
            for (int nt = 0; nt < 2; ++nt) {
                Frag16 b;
                b.h[0] = *(const v8h*)(wb + nt * 16 * WSTRIDE + kt * 32);
                b.h[1] = *(const v8h*)(wb + nt * 16 * WSTRIDE + kt * 32 + 16);
                if (kt & 1)
                    acc1[nt] = __builtin_amdgcn_wmma_f32_16x16x32_f16(
                        false, a.v, false, b.v, (short)0, acc1[nt], false, false);
                else
                    acc0[nt] = __builtin_amdgcn_wmma_f32_16x16x32_f16(
                        false, a.v, false, b.v, (short)0, acc0[nt], false, false);
            }
        }

        // h_new = tanh(xp + h@W^T) -> other buffer (base + immediates)
#pragma unroll
        for (int nt = 0; nt < 2; ++nt)
#pragma unroll
            for (int r = 0; r < 8; ++r) {
                const float v = acc0[nt][r] + acc1[nt][r];
                hOut[(size_t)r * WSTRIDE + nt * 16] = (_Float16)tanh_hw(v);
            }
        __syncthreads();

        if (t + 1 < T_SEQ) {
#pragma unroll
            for (int nt = 0; nt < 2; ++nt)
#pragma unroll
                for (int r = 0; r < 8; ++r) xpv[nt][r] = xn[nt][r];
        }
        xprow = xnrow;
    };

    for (int t = 0; t < T_SEQ; t += 2) {
        step(aA, oB, t);       // read hbA, write hbB
        step(aB, oA, t + 1);   // read hbB, write hbA
    }

    // T_SEQ even -> final h lives in hbA. FC (256->2) + log_softmax.
    float* logits = (float*)hbB;  // reuse buffer B as [16][2] f32
    if (tid < 32) {
        const int l = tid >> 1, o = tid & 1;
        const _Float16* hrow = hbA + (size_t)l * WSTRIDE;
        const float*    w    = W_fc + (size_t)o * HID;
        float s = b_fc[o];
        for (int k = 0; k < HID; ++k) s += (float)hrow[k] * w[k];
        logits[l * 2 + o] = s;
    }
    __syncthreads();
    if (tid < 32) {
        const int l = tid >> 1, o = tid & 1;
        const float l0 = logits[l * 2 + 0], l1 = logits[l * 2 + 1];
        const float m   = fmaxf(l0, l1);
        const float lse = m + __logf(__expf(l0 - m) + __expf(l1 - m));
        out[(size_t)(bRow0 + l) * 2 + o] = logits[l * 2 + o] - lse;
    }
}

extern "C" void kernel_launch(void* const* d_in, const int* in_sizes, int n_in,
                              void* d_out, int out_size, void* d_ws, size_t ws_size,
                              hipStream_t stream) {
    const float* x    = (const float*)d_in[0];
    const float* W_ih = (const float*)d_in[1];
    const float* W_hh = (const float*)d_in[2];
    const float* b_ih = (const float*)d_in[3];
    const float* b_hh = (const float*)d_in[4];
    const float* W_fc = (const float*)d_in[5];
    const float* b_fc = (const float*)d_in[6];
    float*    out = (float*)d_out;
    _Float16* xp  = (_Float16*)d_ws;  // [T*B][HID] f16 = 64 MiB

    const int lds_xproj = HID * KPAD * (int)sizeof(_Float16);                   // 167936 B
    const int lds_scan  = (HID + 2 * 16) * WSTRIDE * (int)sizeof(_Float16);     // 152064 B
    (void)hipFuncSetAttribute(reinterpret_cast<const void*>(&k_xproj),
                              hipFuncAttributeMaxDynamicSharedMemorySize, lds_xproj);
    (void)hipFuncSetAttribute(reinterpret_cast<const void*>(&k_scan),
                              hipFuncAttributeMaxDynamicSharedMemorySize, lds_scan);

    k_xproj<<<512, 256, lds_xproj, stream>>>(x, W_ih, b_ih, b_hh, xp);
    k_scan<<<16, 256, lds_scan, stream>>>(xp, W_hh, W_fc, b_fc, out);
}
